// EncoderBlock_84885733638467
// MI455X (gfx1250) — matrix operand order repaired and checked by reading, not compiled, not run on hardware
//
#include <hip/hip_runtime.h>

// ---------------------------------------------------------------------------
// EncoderBlock for gfx1250 (MI455X).  bf16 WMMA everywhere a matmul appears.
// ---------------------------------------------------------------------------

typedef __attribute__((ext_vector_type(16))) __bf16 v16bf;
typedef __attribute__((ext_vector_type(8)))  float  v8f;
typedef int v4i_ __attribute__((__vector_size__(4 * sizeof(int))));

#if defined(__has_builtin)
#if __has_builtin(__builtin_amdgcn_global_load_async_to_lds_b128) && \
    __has_builtin(__builtin_amdgcn_s_wait_asynccnt)
#define USE_ASYNC_LDS 1
#endif
#endif
#ifndef USE_ASYNC_LDS
#define USE_ASYNC_LDS 0
#endif

union Frag {
    v16bf bf;
    uint4 q[2];
};

constexpr int Bn  = 8;
constexpr int Cn  = 512;
constexpr int Ln  = 1024;
constexpr int Hn  = 8;
constexpr int DHn = 64;
constexpr int Mn  = Bn * Ln;          // 8192 rows for all GEMMs
constexpr int KWn = 7;
constexpr float NEGC = -1e30f;

// ---------------------------------------------------------------------------
// float -> bf16 conversion (weights)
// ---------------------------------------------------------------------------
__global__ void cvt_bf16_kernel(const float* __restrict__ src,
                                __bf16* __restrict__ dst, int n) {
    int i = blockIdx.x * 256 + threadIdx.x;
    if (i < n) dst[i] = (__bf16)src[i];
}

// ---------------------------------------------------------------------------
// cur[b,l,c] = x[b,c,l] + pe(c,l)
// ---------------------------------------------------------------------------
__global__ void posenc_kernel(const float* __restrict__ x,
                              float* __restrict__ cur) {
    int idx = blockIdx.x * 256 + threadIdx.x;      // B*L*C threads
    int c = idx & (Cn - 1);
    int l = (idx >> 9) & (Ln - 1);
    int b = idx >> 19;
    float d   = __expf(-(float)(c & ~1) * (9.210340372f / (float)Cn));
    float ang = (float)l * d;
    float pe  = (c & 1) ? __cosf(ang) : __sinf(ang);
    cur[idx] = x[((size_t)b * Cn + c) * Ln + l] + pe;
}

// ---------------------------------------------------------------------------
// LayerNorm over C=512, one wave32 per row (16 elems/lane).
// ---------------------------------------------------------------------------
__global__ __launch_bounds__(256)
void ln_kernel(const float* __restrict__ in, const float* __restrict__ g,
               const float* __restrict__ bb, float* __restrict__ outf,
               __bf16* __restrict__ outb) {
    const int tid  = threadIdx.x;
    const int lane = tid & 31;
    const int w    = tid >> 5;
    const int row  = blockIdx.x * 8 + w;
    const float* xp = in + (size_t)row * Cn + lane * 16;

    float v[16];
    float s = 0.f;
#pragma unroll
    for (int j = 0; j < 16; ++j) { v[j] = xp[j]; s += v[j]; }
#pragma unroll
    for (int off = 16; off >= 1; off >>= 1) s += __shfl_xor(s, off, 32);
    float mu = s * (1.f / (float)Cn);

    float q = 0.f;
#pragma unroll
    for (int j = 0; j < 16; ++j) { float d = v[j] - mu; q += d * d; }
#pragma unroll
    for (int off = 16; off >= 1; off >>= 1) q += __shfl_xor(q, off, 32);
    float inv = rsqrtf(q * (1.f / (float)Cn) + 1e-5f);

#pragma unroll
    for (int j = 0; j < 16; ++j) {
        int c = lane * 16 + j;
        float y = (v[j] - mu) * inv * g[c] + bb[c];
        size_t o = (size_t)row * Cn + c;
        if (outf) outf[o] = y;
        if (outb) outb[o] = (__bf16)y;
    }
}

// ---------------------------------------------------------------------------
// Depthwise conv along L (K=7, same-pad), (B,L,C) layout, bf16 output.
// ---------------------------------------------------------------------------
__global__ __launch_bounds__(256)
void dwconv_kernel(const float* __restrict__ in, const float* __restrict__ w7,
                   const float* __restrict__ bias, __bf16* __restrict__ out) {
    int idx = blockIdx.x * 256 + threadIdx.x;      // M*C/4 threads
    int c4 = (idx & (Cn / 4 - 1)) * 4;
    int m  = idx >> 7;
    int l  = m & (Ln - 1);
    int b  = m >> 10;

    float acc[4];
#pragma unroll
    for (int j = 0; j < 4; ++j) acc[j] = bias[c4 + j];

    for (int k = 0; k < KWn; ++k) {
        int ll = l + k - KWn / 2;
        if (ll < 0 || ll >= Ln) continue;
        const float* src = in + ((size_t)(b * Ln + ll) * Cn + c4);
#pragma unroll
        for (int j = 0; j < 4; ++j) acc[j] += src[j] * w7[(c4 + j) * KWn + k];
    }
#pragma unroll
    for (int j = 0; j < 4; ++j) out[(size_t)m * Cn + c4 + j] = (__bf16)acc[j];
}

// ---------------------------------------------------------------------------
// GEMM: out[m,n] = act( A[m,:] . W[n,:] + bias[n] ) (+resid).  M=8192, N=K=512.
// 128x128 tile / workgroup; 8 waves of 32x64; WMMA f32_16x16x32_bf16.
// Tile staging uses GLOBAL_LOAD_ASYNC_TO_LDS_B128 (ASYNCcnt) when available.
// ---------------------------------------------------------------------------
constexpr int TM  = 128;
constexpr int TN  = 128;
constexpr int TK  = 32;
constexpr int LDP = 48;   // padded LDS row stride in bf16 elems (96B, 16B mult)

__global__ __launch_bounds__(256)
void gemm_kernel(const __bf16* __restrict__ A, const __bf16* __restrict__ W,
                 const float* __restrict__ bias, const float* __restrict__ resid,
                 float* __restrict__ outf, __bf16* __restrict__ outb, int relu) {
    __shared__ __align__(16) __bf16 Ash[TM * LDP];
    __shared__ __align__(16) __bf16 Bsh[TN * LDP];

    const int tid  = threadIdx.x;
    const int bn   = blockIdx.x;            // N / 128 = 4
    const int bm   = blockIdx.y;            // M / 128 = 64
    const int lane = tid & 31;
    const int w    = tid >> 5;
    const int lo   = lane & 15;
    const int hi   = lane >> 4;
    const int wm   = w >> 1;                // 0..3
    const int wn   = w & 1;                 // 0..1

    v8f zero = {0.f, 0.f, 0.f, 0.f, 0.f, 0.f, 0.f, 0.f};
    v8f acc[2][4];
#pragma unroll
    for (int i = 0; i < 2; ++i)
#pragma unroll
        for (int j = 0; j < 4; ++j) acc[i][j] = zero;

    const int ldr = tid >> 1;               // 0..127
    const int ldk = (tid & 1) * 16;         // 0 / 16

    for (int kk = 0; kk < Cn; kk += TK) {
        // stage A and W tiles into LDS
        {
            const __bf16* sa = A + (size_t)(bm * TM + ldr) * Cn + kk + ldk;
            const __bf16* sb = W + (size_t)(bn * TN + ldr) * Cn + kk + ldk;
            __bf16* da = Ash + ldr * LDP + ldk;
            __bf16* db = Bsh + ldr * LDP + ldk;
#if USE_ASYNC_LDS
            // async DMA path: global -> LDS without VGPR round trip (ASYNCcnt)
            __builtin_amdgcn_global_load_async_to_lds_b128(
                (v4i_*)sa,
                (__attribute__((address_space(3))) v4i_*)da, 0, 0);
            __builtin_amdgcn_global_load_async_to_lds_b128(
                (v4i_*)(sa + 8),
                (__attribute__((address_space(3))) v4i_*)(da + 8), 0, 0);
            __builtin_amdgcn_global_load_async_to_lds_b128(
                (v4i_*)sb,
                (__attribute__((address_space(3))) v4i_*)db, 0, 0);
            __builtin_amdgcn_global_load_async_to_lds_b128(
                (v4i_*)(sb + 8),
                (__attribute__((address_space(3))) v4i_*)(db + 8), 0, 0);
            __builtin_amdgcn_s_wait_asynccnt(0);
#else
            ((uint4*)da)[0] = ((const uint4*)sa)[0];
            ((uint4*)da)[1] = ((const uint4*)sa)[1];
            ((uint4*)db)[0] = ((const uint4*)sb)[0];
            ((uint4*)db)[1] = ((const uint4*)sb)[1];
#endif
        }
        if (kk + TK < Cn)
            __builtin_prefetch(A + (size_t)(bm * TM + ldr) * Cn + kk + TK + ldk, 0, 3);
        __syncthreads();

        Frag af[2], bw[4];
#pragma unroll
        for (int mt = 0; mt < 2; ++mt) {
            const uint4* p = (const uint4*)(Ash + (wm * 32 + mt * 16 + lo) * LDP);
            af[mt].q[0] = p[hi];            // K halves hi*8 .. +7
            af[mt].q[1] = p[2 + hi];        // K halves 16+hi*8 .. +7
        }
#pragma unroll
        for (int nt = 0; nt < 4; ++nt) {
            const uint4* p = (const uint4*)(Bsh + (wn * 64 + nt * 16 + lo) * LDP);
            bw[nt].q[0] = p[hi];            // K halves hi*8 .. +7
            bw[nt].q[1] = p[2 + hi];        // K halves 16+hi*8 .. +7
        }
#pragma unroll
        for (int mt = 0; mt < 2; ++mt)
#pragma unroll
            for (int nt = 0; nt < 4; ++nt)
                acc[mt][nt] = __builtin_amdgcn_wmma_f32_16x16x32_bf16(
                    false, af[mt].bf, false, bw[nt].bf, (short)0, acc[mt][nt],
                    false, false);
        __syncthreads();
    }

    // epilogue: C layout -> m = r + 8*hi, n = lo (within 16x16 tile)
#pragma unroll
    for (int mt = 0; mt < 2; ++mt)
#pragma unroll
        for (int nt = 0; nt < 4; ++nt)
#pragma unroll
            for (int r = 0; r < 8; ++r) {
                int grow = bm * TM + wm * 32 + mt * 16 + r + 8 * hi;
                int gcol = bn * TN + wn * 64 + nt * 16 + lo;
                float v = acc[mt][nt][r] + bias[gcol];
                if (relu) v = fmaxf(v, 0.f);
                size_t o = (size_t)grow * Cn + gcol;
                if (resid) v += resid[o];
                if (outf) outf[o] = v;
                if (outb) outb[o] = (__bf16)v;
            }
}

// ---------------------------------------------------------------------------
// V transpose: [B,L,H*DH] -> [B,H,DH,L] so P@V fragments are contiguous.
// ---------------------------------------------------------------------------
__global__ void vtr_kernel(const __bf16* __restrict__ v, __bf16* __restrict__ vT) {
    int idx = blockIdx.x * 256 + threadIdx.x;      // B*H*DH*L threads
    int l  = idx & (Ln - 1);
    int dh = (idx >> 10) & (DHn - 1);
    int h  = (idx >> 16) & (Hn - 1);
    int b  = idx >> 19;
    vT[idx] = v[((size_t)(b * Ln + l) * Cn) + h * DHn + dh];
}

// ---------------------------------------------------------------------------
// Fused attention: per (b,h,16-row Q block).  Two-pass softmax (scores are
// recomputed instead of materialized).  Q/K/V fragments from global.
// ---------------------------------------------------------------------------
__device__ inline void smmerge(float& m, float& s, float om, float os) {
    float mn = fmaxf(m, om);
    s = s * __expf(m - mn) + os * __expf(om - mn);
    m = mn;
}

__global__ __launch_bounds__(256)
void attn_kernel(const __bf16* __restrict__ Q, const __bf16* __restrict__ Kb,
                 const __bf16* __restrict__ Vt, const float* __restrict__ mask,
                 __bf16* __restrict__ Oout) {
    __shared__ float wsm[8 * 16], wss[8 * 16];
    __shared__ float fm[16], fs[16];
    __shared__ __align__(16) __bf16 Psh[8][16 * LDP];
    __shared__ float Obuf[16 * 64];

    const int tid  = threadIdx.x;
    const int lane = tid & 31;
    const int w    = tid >> 5;       // wave owns 32-key strip inside 256-key blk
    const int lo   = lane & 15;
    const int hi   = lane >> 4;
    const int qb   = blockIdx.x & 63;
    const int h    = (blockIdx.x >> 6) & 7;
    const int b    = blockIdx.x >> 9;
    const float scale = 0.125f;      // 1/sqrt(64)

    // Q A-fragments (rows qb*16+lo, dh contiguous in global)
    Frag qa0, qa1;
    {
        const uint4* p = (const uint4*)(Q + ((size_t)(b * Ln + qb * 16 + lo) * Cn + h * DHn));
        qa0.q[0] = p[hi];     qa0.q[1] = p[2 + hi];    // dh 0..31
        qa1.q[0] = p[4 + hi]; qa1.q[1] = p[6 + hi];    // dh 32..63
    }

    v8f zero = {0.f, 0.f, 0.f, 0.f, 0.f, 0.f, 0.f, 0.f};

    // ---------------- pass A: row max / sum of exp ----------------
    float mrun[8], srun[8];
#pragma unroll
    for (int r = 0; r < 8; ++r) { mrun[r] = -3e38f; srun[r] = 0.f; }

    for (int kb = 0; kb < Ln; kb += 256) {
#pragma unroll
        for (int t = 0; t < 2; ++t) {
            const int l0 = kb + w * 32 + t * 16;
            Frag kf0, kf1;
            const uint4* p = (const uint4*)(Kb + ((size_t)(b * Ln + l0 + lo) * Cn + h * DHn));
            kf0.q[0] = p[hi];     kf0.q[1] = p[2 + hi];       // dh 0..31
            kf1.q[0] = p[4 + hi]; kf1.q[1] = p[6 + hi];   // dh 32..63
            v8f acc = zero;
            acc = __builtin_amdgcn_wmma_f32_16x16x32_bf16(false, qa0.bf, false, kf0.bf,
                                                          (short)0, acc, false, false);
            acc = __builtin_amdgcn_wmma_f32_16x16x32_bf16(false, qa1.bf, false, kf1.bf,
                                                          (short)0, acc, false, false);
            const float mv = mask[(size_t)b * Ln + l0 + lo];
#pragma unroll
            for (int r = 0; r < 8; ++r) {
                float sv = acc[r] * scale;
                sv = sv * mv + (1.f - mv) * NEGC;
                smmerge(mrun[r], srun[r], sv, 1.f);
            }
        }
    }
    // intra-wave reduce across the 16 column lanes of each half
#pragma unroll
    for (int off = 1; off < 16; off <<= 1)
#pragma unroll
        for (int r = 0; r < 8; ++r) {
            float om = __shfl_xor(mrun[r], off, 32);
            float os = __shfl_xor(srun[r], off, 32);
            smmerge(mrun[r], srun[r], om, os);
        }
    if (lo == 0)
#pragma unroll
        for (int r = 0; r < 8; ++r) {
            wsm[w * 16 + hi * 8 + r] = mrun[r];
            wss[w * 16 + hi * 8 + r] = srun[r];
        }
    __syncthreads();
    if (tid < 16) {
        float m = -3e38f, s = 0.f;
        for (int w2 = 0; w2 < 8; ++w2) smmerge(m, s, wsm[w2 * 16 + tid], wss[w2 * 16 + tid]);
        fm[tid] = m;
        fs[tid] = (s > 0.f) ? (1.f / s) : 0.f;
    }
    __syncthreads();

    // ---------------- pass B: P = softmax(S); O += P @ V ----------------
    v8f accO[4];
#pragma unroll
    for (int nt = 0; nt < 4; ++nt) accO[nt] = zero;

    for (int kb = 0; kb < Ln; kb += 256) {
#pragma unroll
        for (int t = 0; t < 2; ++t) {
            const int l0 = kb + w * 32 + t * 16;
            Frag kf0, kf1;
            const uint4* p = (const uint4*)(Kb + ((size_t)(b * Ln + l0 + lo) * Cn + h * DHn));
            kf0.q[0] = p[hi];     kf0.q[1] = p[2 + hi];
            kf1.q[0] = p[4 + hi]; kf1.q[1] = p[6 + hi];
            v8f acc = zero;
            acc = __builtin_amdgcn_wmma_f32_16x16x32_bf16(false, qa0.bf, false, kf0.bf,
                                                          (short)0, acc, false, false);
            acc = __builtin_amdgcn_wmma_f32_16x16x32_bf16(false, qa1.bf, false, kf1.bf,
                                                          (short)0, acc, false, false);
            const float mv = mask[(size_t)b * Ln + l0 + lo];
#pragma unroll
            for (int r = 0; r < 8; ++r) {
                float sv = acc[r] * scale;
                sv = sv * mv + (1.f - mv) * NEGC;
                int row16 = r + 8 * hi;
                float pv = __expf(sv - fm[row16]) * fs[row16];
                Psh[w][row16 * LDP + t * 16 + lo] = (__bf16)pv;
            }
        }
        // LDS is in-order per wave; fence the compiler + hardware anyway
        asm volatile("s_wait_dscnt 0" ::: "memory");

        Frag pf;
        {
            const uint4* pp = (const uint4*)(&Psh[w][lo * LDP]);
            pf.q[0] = pp[2 * hi];       // k halves hi*16 .. +7  (keys kb+w*32+..)
            pf.q[1] = pp[2 * hi + 1];   // k halves hi*16+8 .. +15
        }
#pragma unroll
        for (int nt = 0; nt < 4; ++nt) {
            Frag vf;
            const uint4* vp = (const uint4*)(Vt +
                ((size_t)((b * Hn + h) * DHn + nt * 16 + lo) * Ln + kb + w * 32 + hi * 16));
            vf.q[0] = vp[0];
            vf.q[1] = vp[1];
            accO[nt] = __builtin_amdgcn_wmma_f32_16x16x32_bf16(
                false, pf.bf, false, vf.bf, (short)0, accO[nt], false, false);
        }
    }

    // cross-wave O reduction
#pragma unroll
    for (int i = 0; i < 4; ++i) Obuf[tid * 4 + i] = 0.f;
    __syncthreads();
#pragma unroll
    for (int nt = 0; nt < 4; ++nt)
#pragma unroll
        for (int r = 0; r < 8; ++r)
            atomicAdd(&Obuf[(r + 8 * hi) * 64 + nt * 16 + lo], accO[nt][r]);
    __syncthreads();
#pragma unroll
    for (int i = 0; i < 4; ++i) {
        int e   = tid * 4 + i;
        int row = e >> 6;
        int dh  = e & 63;
        Oout[((size_t)(b * Ln + qb * 16 + row) * Cn) + h * DHn + dh] = (__bf16)Obuf[e];
    }
}

// ---------------------------------------------------------------------------
// Host orchestration
// ---------------------------------------------------------------------------
extern "C" void kernel_launch(void* const* d_in, const int* in_sizes, int n_in,
                              void* d_out, int out_size, void* d_ws, size_t ws_size,
                              hipStream_t stream) {
    (void)in_sizes; (void)n_in; (void)out_size; (void)ws_size;

    const float* x        = (const float*)d_in[0];
    const float* mask     = (const float*)d_in[1];
    const float* dw_w     = (const float*)d_in[2];
    const float* dw_b     = (const float*)d_in[3];
    const float* pw_w     = (const float*)d_in[4];
    const float* pw_b     = (const float*)d_in[5];
    const float* lncg     = (const float*)d_in[6];
    const float* lncb     = (const float*)d_in[7];
    const float* Wq       = (const float*)d_in[8];
    const float* bq       = (const float*)d_in[9];
    const float* Wk       = (const float*)d_in[10];
    const float* bk       = (const float*)d_in[11];
    const float* Wv       = (const float*)d_in[12];
    const float* bv       = (const float*)d_in[13];
    const float* Wo       = (const float*)d_in[14];
    const float* bo       = (const float*)d_in[15];
    const float* lnag     = (const float*)d_in[16];
    const float* lnab     = (const float*)d_in[17];
    const float* Wf       = (const float*)d_in[18];
    const float* bff      = (const float*)d_in[19];
    const float* lnfg     = (const float*)d_in[20];
    const float* lnfb     = (const float*)d_in[21];
    float* out = (float*)d_out;

    const size_t CC = (size_t)Cn * Cn;
    char* ws = (char*)d_ws;
    size_t off = 0;
    float*  cur   = (float*)(ws + off);  off += (size_t)Mn * Cn * 4;   // 16 MB
    float*  lnout = (float*)(ws + off);  off += (size_t)Mn * Cn * 4;   // 16 MB
    __bf16* Abf   = (__bf16*)(ws + off); off += (size_t)Mn * Cn * 2;   //  8 MB
    __bf16* Wbf   = (__bf16*)(ws + off); off += 9 * CC * 2;            // 4.5 MB
    __bf16* qbf   = (__bf16*)(ws + off); off += (size_t)Mn * Cn * 2;
    __bf16* kbf   = (__bf16*)(ws + off); off += (size_t)Mn * Cn * 2;
    __bf16* vbf   = (__bf16*)(ws + off); off += (size_t)Mn * Cn * 2;
    __bf16* vT    = (__bf16*)(ws + off); off += (size_t)Mn * Cn * 2;   // ~76.5 MB total

    // weights -> bf16 (pw_w 4 matrices, then Wq Wk Wv Wo Wf)
    cvt_bf16_kernel<<<(int)((4 * CC + 255) / 256), 256, 0, stream>>>(pw_w, Wbf, (int)(4 * CC));
    cvt_bf16_kernel<<<(int)((CC + 255) / 256), 256, 0, stream>>>(Wq, Wbf + 4 * CC, (int)CC);
    cvt_bf16_kernel<<<(int)((CC + 255) / 256), 256, 0, stream>>>(Wk, Wbf + 5 * CC, (int)CC);
    cvt_bf16_kernel<<<(int)((CC + 255) / 256), 256, 0, stream>>>(Wv, Wbf + 6 * CC, (int)CC);
    cvt_bf16_kernel<<<(int)((CC + 255) / 256), 256, 0, stream>>>(Wo, Wbf + 7 * CC, (int)CC);
    cvt_bf16_kernel<<<(int)((CC + 255) / 256), 256, 0, stream>>>(Wf, Wbf + 8 * CC, (int)CC);

    posenc_kernel<<<Mn * Cn / 256, 256, 0, stream>>>(x, cur);

    dim3 ggrid(Cn / TN, Mn / TM);   // (4, 64)

    // 4 conv blocks
    for (int i = 0; i < 4; ++i) {
        ln_kernel<<<Mn / 8, 256, 0, stream>>>(cur, lncg + i * Cn, lncb + i * Cn,
                                              lnout, (__bf16*)nullptr);
        dwconv_kernel<<<Mn * Cn / 4 / 256, 256, 0, stream>>>(lnout, dw_w + (size_t)i * Cn * KWn,
                                                             dw_b + i * Cn, Abf);
        gemm_kernel<<<ggrid, 256, 0, stream>>>(Abf, Wbf + (size_t)i * CC, pw_b + i * Cn,
                                               cur, cur, (__bf16*)nullptr, 1);
    }

    // attention
    ln_kernel<<<Mn / 8, 256, 0, stream>>>(cur, lnag, lnab, (float*)nullptr, Abf);
    gemm_kernel<<<ggrid, 256, 0, stream>>>(Abf, Wbf + 4 * CC, bq, (const float*)nullptr,
                                           (float*)nullptr, qbf, 0);
    gemm_kernel<<<ggrid, 256, 0, stream>>>(Abf, Wbf + 5 * CC, bk, (const float*)nullptr,
                                           (float*)nullptr, kbf, 0);
    gemm_kernel<<<ggrid, 256, 0, stream>>>(Abf, Wbf + 6 * CC, bv, (const float*)nullptr,
                                           (float*)nullptr, vbf, 0);
    vtr_kernel<<<Mn * Cn / 256, 256, 0, stream>>>(vbf, vT);
    attn_kernel<<<Bn * Hn * (Ln / 16), 256, 0, stream>>>(qbf, kbf, vT, mask, Abf);
    gemm_kernel<<<ggrid, 256, 0, stream>>>(Abf, Wbf + 7 * CC, bo, cur, cur,
                                           (__bf16*)nullptr, 0);

    // FFN (writes final output)
    ln_kernel<<<Mn / 8, 256, 0, stream>>>(cur, lnfg, lnfb, (float*)nullptr, Abf);
    gemm_kernel<<<ggrid, 256, 0, stream>>>(Abf, Wbf + 8 * CC, bff, cur, out,
                                           (__bf16*)nullptr, 1);
}
